// Selected_33835752357929
// MI455X (gfx1250) — compile-verified, matrix operand order checked by reading, not run on hardware
//
#include <hip/hip_runtime.h>
#include <math.h>

typedef __attribute__((ext_vector_type(16))) _Float16 v16h;
typedef __attribute__((ext_vector_type(8)))  float    v8f;

#define WAVES 8

__device__ inline float sigm(float x){ return 1.f/(1.f+expf(-x)); }

// ---------------- WMMA block GEMM ----------------
// C[r][n] = act( sum_p A_p[rowBase+r][k] * W_p[n][wofs+k] + bias[n] + addv[r][n] + rowvec[(rowBase+r)/rowdiv][n] )
// W is f16, row-major N x ldw.  Executed by a 256-thread block (8 wave32s).
struct Part { const float* A; long lda; const _Float16* W; long ldw; int wofs; int K; };

__device__ void gemm_block(const Part* parts, int np,
                           long rowBase, int Mvalid, int N,
                           const float* bias,
                           const float* addv, long ldadd,
                           const float* rowvec, int rowdiv,
                           int act,              // 0 none, 1 relu, 2 tanh
                           float* C, long ldc)
{
  int tid  = threadIdx.x;
  int wave = tid >> 5;
  int lane = tid & 31;
  int mT = (Mvalid + 15) >> 4; if (mT > 4) mT = 4;
  int nT = N >> 4;
  int ntiles = mT * nT;
  for (int t = wave; t < ntiles; t += WAVES) {
    int m0 = (t / nT) << 4;
    int n0 = (t % nT) << 4;
    v8f acc = {};
    for (int p = 0; p < np; ++p) {
      const Part pp = parts[p];
      int r = m0 + (lane & 15);
      if (r >= Mvalid) r = Mvalid - 1;
      const float* abase = pp.A + (rowBase + (long)r) * pp.lda;
      const _Float16* wbase = pp.W + (long)(n0 + (lane & 15)) * pp.ldw + pp.wofs;
      int kh = (lane >> 4) << 3;
      for (int k0 = 0; k0 < pp.K; k0 += 32) {
        const float*    arow = abase + k0;
        const _Float16* wrow = wbase + k0;
        v16h af, bf;
        #pragma unroll
        for (int j = 0; j < 8; ++j) af[j]     = (_Float16)arow[kh + j];
        #pragma unroll
        for (int j = 0; j < 8; ++j) af[8 + j] = (_Float16)arow[16 + kh + j];
        #pragma unroll
        for (int j = 0; j < 8; ++j) bf[j]     = wrow[kh + j];
        #pragma unroll
        for (int j = 0; j < 8; ++j) bf[8 + j] = wrow[16 + kh + j];
        acc = __builtin_amdgcn_wmma_f32_16x16x32_f16(false, af, false, bf,
                                                     (short)0, acc, false, false);
      }
    }
    int col   = n0 + (lane & 15);
    int rbase = m0 + ((lane >> 4) << 3);
    #pragma unroll
    for (int v = 0; v < 8; ++v) {
      int r = rbase + v;
      if (r < Mvalid) {
        float x = acc[v];
        if (bias)   x += bias[col];
        if (addv)   x += addv[(long)r * ldadd + col];
        if (rowvec) x += rowvec[(long)((rowBase + r) / rowdiv) * N + col];
        if (act == 1) x = fmaxf(x, 0.f);
        else if (act == 2) x = tanhf(x);
        C[(long)r * ldc + col] = x;
      }
    }
  }
}

// LayerNorm of 64 rows x 256 cols (blockDim=256). s_red must hold 512 floats.
__device__ void ln64x256(const float* X, float* Y, const float* g, const float* be,
                         float* s_red, int tid)
{
  int r = tid >> 2, q = tid & 3;
  const float* x = X + (long)r * 256;
  float s = 0.f, s2 = 0.f;
  for (int j = q * 64; j < q * 64 + 64; ++j) { float v = x[j]; s += v; s2 += v * v; }
  s_red[r * 8 + q] = s; s_red[r * 8 + 4 + q] = s2;
  __syncthreads();
  if (q == 0) {
    float ts = s_red[r*8] + s_red[r*8+1] + s_red[r*8+2] + s_red[r*8+3];
    float t2 = s_red[r*8+4] + s_red[r*8+5] + s_red[r*8+6] + s_red[r*8+7];
    float m  = ts * (1.f/256.f);
    float var = t2 * (1.f/256.f) - m * m;
    s_red[r*8]   = m;
    s_red[r*8+1] = rsqrtf(var + 1e-5f);
  }
  __syncthreads();
  for (int pcs = tid; pcs < 64 * 256; pcs += 256) {
    int rr = pcs >> 8, j = pcs & 255;
    float m = s_red[rr*8], rs = s_red[rr*8+1];
    Y[pcs] = (X[pcs] - m) * rs * g[j] + be[j];
  }
  __syncthreads();
}

// ---------------- kernels ----------------
__global__ void k_h(const float* s, _Float16* d, long n) {
  long i = (long)blockIdx.x * 256 + threadIdx.x;
  if (i < n) d[i] = (_Float16)s[i];
}

struct GemmArgs {
  Part p[3]; int np;
  long rows; int N;
  const float* bias;
  const float* addv; long ldadd;
  const float* rowvec; int rowdiv;
  int act;
  float* C; long ldc;
};
__global__ void __launch_bounds__(256) k_gemm(GemmArgs g) {
  long rb = (long)blockIdx.x * 64;
  long rem = g.rows - rb;
  int mv = (int)(rem < 64 ? rem : 64);
  gemm_block(g.p, g.np, rb, mv, g.N, g.bias,
             g.addv ? g.addv + rb * g.ldadd : nullptr, g.ldadd,
             g.rowvec, g.rowdiv, g.act, g.C + rb * g.ldc, g.ldc);
}

struct PrepArgs {
  const float* betas; const float* tgt;
  const float* tl_w; const float* tl_b; const float* tl2_w; const float* tl2_b;
  const float* cs1_w; const float* cs1_b;
  const float* vs1_w; const float* vs1_b;
  const float* cq_w;  const float* cq_b;
  const float* ga1_w; const float* ga1_b;
  const float* hl_bih; const float* hl_bhh;
  float* cp; float* tlin; float* tlin2; float* csconst; float* vconst;
  float* cqconst; float* gaconst; float* hlb; float* accum;
};
__global__ void __launch_bounds__(256) k_prep(PrepArgs a) {
  int tid = threadIdx.x;
  if (tid == 0) {
    float c = 1.f;
    for (int t = 0; t < 1000; ++t) { c *= (1.f - a.betas[t]); a.cp[t] = c; }
    a.accum[0] = 0.f; a.accum[1] = 0.f;
  }
  {
    int n = tid;
    float s1 = a.tl_b[n], s2 = a.tl2_b[n], s3 = a.cs1_b[n], s4 = a.ga1_b[n];
    for (int d = 0; d < 256; ++d) {
      float tg = a.tgt[d];
      s1 += tg * a.tl_w[(long)n*256 + d];
      s2 += tg * a.tl2_w[(long)n*256 + d];
      s3 += tg * a.cs1_w[(long)n*512 + 256 + d];
      s4 += tg * a.ga1_w[(long)n*576 + 320 + d];
    }
    a.tlin[n]=s1; a.tlin2[n]=s2; a.csconst[n]=s3; a.gaconst[n]=s4;
  }
  for (int j = tid; j < 1024; j += 256) a.hlb[j] = a.hl_bih[j] + a.hl_bhh[j];
  __syncthreads();
  {
    int n = tid;
    float s1 = a.vs1_b[n], s2 = a.cq_b[n];
    for (int d = 0; d < 256; ++d) {
      s1 += a.tlin2[d] * a.vs1_w[(long)n*800 + 544 + d];
      s2 += a.tlin[d]  * a.cq_w[(long)n*544 + 288 + d];
    }
    a.vconst[n]=s1; a.cqconst[n]=s2;
  }
}

__global__ void k_time(const float* stime, const float* t1_w, const float* t1_b,
                       const float* t2_w, const float* t2_b, float* tf, float* tf2) {
  long i = (long)blockIdx.x * 256 + threadIdx.x;
  if (i < 3200L * 64) {
    int r = (int)(i >> 6), k = (int)(i & 63);
    float ts = stime[r] * (1.f/180.f);
    float y1 = ts * t1_w[k] + t1_b[k];
    float y2 = ts * t2_w[k] + t2_b[k];
    tf[i]  = 1.f - tanhf(y1 * y1);
    tf2[i] = 1.f - tanhf(y2 * y2);
  }
}

struct PmArgs { const float* emb; const _Float16* h_cs1; const float* csconst;
  const float* cs2_w; const float* cs2_b; const float* gum; float* pm; };
__global__ void __launch_bounds__(256) k_pm(PmArgs a) {
  __shared__ float h[32 * 256];
  long rb = (long)blockIdx.x * 32;
  long rows = 20001;
  long rem = rows - rb;
  int mv = (int)(rem < 32 ? rem : 32);
  Part p = { a.emb, 256, a.h_cs1, 512, 0, 256 };
  gemm_block(&p, 1, rb, mv, 256, a.csconst, nullptr, 0, nullptr, 0, 2, h, 256);
  __syncthreads();
  int tid = threadIdx.x;
  if (tid < mv) {
    const float* hr = h + tid * 256;
    float l0 = a.cs2_b[0], l1 = a.cs2_b[1];
    for (int n = 0; n < 256; ++n) { l0 += hr[n] * a.cs2_w[n]; l1 += hr[n] * a.cs2_w[256 + n]; }
    long r = rb + tid;
    float v = (l0 + a.gum[r*2] >= l1 + a.gum[r*2+1]) ? 1.f : 0.f;
    if (r == 20000) v = 0.f;
    a.pm[r] = v;
  }
}

__global__ void __launch_bounds__(256) k_selected(const int* idx, const float* emb,
                        const float* pm, const float* te, float* sel) {
  __shared__ int   code[40];
  __shared__ float pmv[40];
  long row = blockIdx.x;               // b*50+s
  int tid = threadIdx.x;
  if (tid < 40) { int c = idx[row * 40 + tid]; code[tid] = c; pmv[tid] = pm[c]; }
  __syncthreads();
  int d = tid;
  float s = te[row * 256 + d];
  for (int c = 0; c < 40; ++c) s += pmv[c] * fmaxf(emb[(long)code[c] * 256 + d], 0.f);
  sel[row * 256 + d] = s;
}

struct FolArgs { const float* selected; const _Float16* h_wih; const _Float16* h_whh;
  float* fh; float* fcs; float* fg; float* fol; };
__global__ void __launch_bounds__(256) k_fol(FolArgs a) {
  int tid = threadIdx.x;
  for (int i = tid; i < 64 * 32; i += 256) { a.fh[i] = 0.f; a.fcs[i] = 0.f; }
  __syncthreads();
  for (int t = 49; t >= 0; --t) {
    Part p[2] = {
      { a.selected + (long)t * 256, 50L * 256, a.h_wih, 256, 0, 256 },
      { a.fh, 32, a.h_whh, 32, 0, 32 } };
    gemm_block(p, 2, 0, 64, 128, nullptr, nullptr, 0, nullptr, 0, 0, a.fg, 128);
    __syncthreads();
    for (int pcs = tid; pcs < 64 * 32; pcs += 256) {
      int b = pcs >> 5, j = pcs & 31;
      const float* g = a.fg + (long)b * 128;
      float c = sigm(g[32 + j]) * a.fcs[pcs] + sigm(g[j]) * tanhf(g[64 + j]);
      float h = sigm(g[96 + j]) * tanhf(c);
      a.fcs[pcs] = c; a.fh[pcs] = h;
      a.fol[((long)b * 50 + t) * 32 + j] = h;
    }
    __syncthreads();
  }
}

struct ScanArgs {
  const float* emb; const int* idx; const float* gum_v; const float* pm;
  const float* selected; const float* fol; const float* time_enc;
  const _Float16* h_vs1; const _Float16* h_cq; const _Float16* h_fc;
  const _Float16* h_gru_wih; const _Float16* h_gru_whh;
  const _Float16* h_mem; const _Float16* h_st; const _Float16* h_ff1; const _Float16* h_ff2;
  const float* vconst; const float* cqconst;
  const float* vs2_w; const float* vs2_b;
  const float* fc_b; const float* mem_b; const float* mem_g; const float* mem_be;
  const float* st_b; const float* st_g; const float* st_be;
  const float* ff1_b; const float* ff2_b;
  const float* ln_g; const float* ln_b; const float* ln2_g; const float* ln2_b;
  float* hx; float* h1; float* cq; float* z; float* zf; float* gi; float* gh;
  float* sh; float* shl; float* hxn; float* hxn2; float* f1;
  float* hiddens; float* memrows; float* accum;
};
__global__ void __launch_bounds__(256) k_scan(ScanArgs a) {
  __shared__ float s_attn[64 * 40];
  __shared__ float s_pmv[64 * 40];
  __shared__ int   s_code[64 * 40];
  __shared__ float s_me[64 * 50];
  __shared__ float s_red[512];
  __shared__ float s_vflag[64];
  __shared__ float s_skip[1];
  int tid = threadIdx.x;
  for (int i = tid; i < 64 * 256; i += 256) a.hx[i] = 0.f;
  if (tid == 0) s_skip[0] = 0.f;
  __syncthreads();
  for (int step = 0; step < 50; ++step) {
    // h1 = tanh(vconst + hx@Va + sel@Vb + fol@Vc)
    {
      Part p[3] = {
        { a.hx, 256, a.h_vs1, 800, 0, 256 },
        { a.selected + (long)step * 256, 50L * 256, a.h_vs1, 800, 256, 256 },
        { a.fol + (long)step * 32, 50L * 32, a.h_vs1, 800, 512, 32 } };
      gemm_block(p, 3, 0, 64, 256, a.vconst, nullptr, 0, nullptr, 0, 2, a.h1, 256);
    }
    __syncthreads();
    if (tid < 64) {    // hard gumbel of vs2 logits
      const float* hr = a.h1 + (long)tid * 256;
      float l0 = a.vs2_b[0], l1 = a.vs2_b[1];
      for (int n = 0; n < 256; ++n) { l0 += hr[n] * a.vs2_w[n]; l1 += hr[n] * a.vs2_w[256 + n]; }
      const float* g = a.gum_v + ((long)step * 64 + tid) * 2;
      s_vflag[tid] = (l0 + g[0] >= l1 + g[1]) ? 1.f : 0.f;
    }
    // cq = relu(cqconst + hx@Ca + fol@Cb)
    {
      Part p[2] = {
        { a.hx, 256, a.h_cq, 544, 0, 256 },
        { a.fol + (long)step * 32, 50L * 32, a.h_cq, 544, 256, 32 } };
      gemm_block(p, 2, 0, 64, 256, a.cqconst, nullptr, 0, nullptr, 0, 1, a.cq, 256);
    }
    __syncthreads();
    // attention over C=40 codes
    for (int pcs = tid; pcs < 64 * 40; pcs += 256) {
      int b = pcs / 40, c = pcs % 40;
      int code = a.idx[((long)b * 50 + step) * 40 + c];
      s_code[pcs] = code;
      const float* er = a.emb + (long)code * 256;
      const float* cr = a.cq + (long)b * 256;
      float e = 0.f;
      for (int d = 0; d < 256; ++d) e += cr[d] * fmaxf(er[d], 0.f);
      s_attn[pcs] = e * 0.0625f;
      s_pmv[pcs] = a.pm[code];
    }
    __syncthreads();
    if (tid < 64) {
      float* e = s_attn + tid * 40;
      const float* pv = s_pmv + tid * 40;
      float m = -1e30f;
      for (int c = 0; c < 40; ++c) m = fmaxf(m, e[c]);
      float ss = 0.f;
      for (int c = 0; c < 40; ++c) { float t = expf(e[c] - m); e[c] = t; ss += t; }
      float inv = 1.f / ss; float pmsum = 0.f;
      for (int c = 0; c < 40; ++c) { e[c] = e[c] * inv * pv[c]; pmsum += pv[c]; }
      s_red[tid] = s_vflag[tid] * pmsum;
    }
    __syncthreads();
    if (tid == 0) { float t = 0.f; for (int b = 0; b < 64; ++b) t += s_red[b]; s_skip[0] += t; }
    for (int pcs = tid; pcs < 64 * 256; pcs += 256) {
      int b = pcs >> 8, d = pcs & 255;
      float zv = 0.f;
      const float* e = s_attn + b * 40;
      const int* cd = s_code + b * 40;
      for (int c = 0; c < 40; ++c) zv += e[c] * fmaxf(a.emb[(long)cd[c] * 256 + d], 0.f);
      a.z[pcs] = zv;
    }
    __syncthreads();
    // zf = z@fc + fc_b + te_i
    {
      Part p[1] = { { a.z, 256, a.h_fc, 256, 0, 256 } };
      gemm_block(p, 1, 0, 64, 256, a.fc_b, a.time_enc + (long)step * 256, 50L * 256,
                 nullptr, 0, 0, a.zf, 256);
    }
    __syncthreads();
    // GRU
    {
      Part p1[1] = { { a.zf, 256, a.h_gru_wih, 256, 0, 256 } };
      gemm_block(p1, 1, 0, 64, 768, nullptr, nullptr, 0, nullptr, 0, 0, a.gi, 768);
      Part p2[1] = { { a.hx, 256, a.h_gru_whh, 256, 0, 256 } };
      gemm_block(p2, 1, 0, 64, 768, nullptr, nullptr, 0, nullptr, 0, 0, a.gh, 768);
    }
    __syncthreads();
    for (int pcs = tid; pcs < 64 * 256; pcs += 256) {
      int b = pcs >> 8, j = pcs & 255;
      const float* gi = a.gi + (long)b * 768;
      const float* gh = a.gh + (long)b * 768;
      float r  = sigm(gi[j] + gh[j]);
      float zz = sigm(gi[256 + j] + gh[256 + j]);
      float nn = tanhf(gi[512 + j] + r * gh[512 + j]);
      float hxv = a.hx[pcs];
      float shv = (1.f - zz) * nn + zz * hxv;
      a.sh[pcs] = (s_vflag[b] > 0.5f) ? shv : hxv;
    }
    __syncthreads();
    const float* hxnew;
    if (step == 0) {
      hxnew = a.sh;
    } else {
      { Part p[1] = { { a.sh, 256, a.h_st, 256, 0, 256 } };
        gemm_block(p, 1, 0, 64, 256, a.st_b, nullptr, 0, nullptr, 0, 1, a.h1, 256); }
      __syncthreads();
      ln64x256(a.h1, a.shl, a.st_g, a.st_be, s_red, tid);
      int ns = step;
      for (int pcs = tid; pcs < 64 * ns; pcs += 256) {
        int b = pcs % 64, s = pcs / 64;
        const float* sr = a.shl + (long)b * 256;
        const float* mr = a.memrows + ((long)s * 64 + b) * 256;
        float e = 0.f;
        for (int d = 0; d < 256; ++d) e += sr[d] * mr[d];
        s_me[b * 50 + s] = e * 0.0625f;
      }
      __syncthreads();
      if (tid < 64) {
        float* r = s_me + tid * 50;
        float m = -1e30f; for (int s = 0; s < ns; ++s) m = fmaxf(m, r[s]);
        float ss = 0.f; for (int s = 0; s < ns; ++s) { float t = expf(r[s] - m); r[s] = t; ss += t; }
        float inv = 1.f / ss; for (int s = 0; s < ns; ++s) r[s] *= inv;
      }
      __syncthreads();
      for (int pcs = tid; pcs < 64 * 256; pcs += 256) {
        int b = pcs >> 8, d = pcs & 255;
        float acc = 0.f;
        const float* w = s_me + b * 50;
        for (int s = 0; s < ns; ++s) acc += w[s] * a.memrows[((long)s * 64 + b) * 256 + d];
        a.z[pcs] = acc + a.shl[pcs];
      }
      __syncthreads();
      ln64x256(a.z, a.hxn, a.ln2_g, a.ln2_b, s_red, tid);
      { Part p[1] = { { a.hxn, 256, a.h_ff1, 256, 0, 256 } };
        gemm_block(p, 1, 0, 64, 1024, a.ff1_b, nullptr, 0, nullptr, 0, 1, a.f1, 1024); }
      __syncthreads();
      { Part p[1] = { { a.f1, 1024, a.h_ff2, 1024, 0, 1024 } };
        gemm_block(p, 1, 0, 64, 256, a.ff2_b, a.hxn, 256, nullptr, 0, 0, a.h1, 256); }
      __syncthreads();
      ln64x256(a.h1, a.hxn2, a.ln_g, a.ln_b, s_red, tid);
      hxnew = a.hxn2;
    }
    for (int pcs = tid; pcs < 64 * 256; pcs += 256) {
      float v = hxnew[pcs];
      a.hx[pcs] = v;
      a.hiddens[(((long)(pcs >> 8)) * 50 + step) * 256 + (pcs & 255)] = v;
    }
    __syncthreads();
    // incremental memory row for this step
    { Part p[1] = { { a.hx, 256, a.h_mem, 256, 0, 256 } };
      gemm_block(p, 1, 0, 64, 256, a.mem_b, nullptr, 0, nullptr, 0, 1, a.h1, 256); }
    __syncthreads();
    ln64x256(a.h1, a.memrows + (long)step * 64 * 256, a.mem_g, a.mem_be, s_red, tid);
  }
  if (tid == 0) a.accum[0] = s_skip[0];
}

struct HlArgs { const float* hiddens; const _Float16* h_wih; const _Float16* h_whh;
  const float* hlb; float* lh; float* lc; float* lg; float* hstate; };
__global__ void __launch_bounds__(256) k_hlstm(HlArgs a) {
  int tid = threadIdx.x;
  for (int i = tid; i < 64 * 256; i += 256) { a.lh[i] = 0.f; a.lc[i] = 0.f; }
  __syncthreads();
  for (int t = 0; t < 50; ++t) {
    Part p[2] = {
      { a.hiddens + (long)t * 256, 50L * 256, a.h_wih, 256, 0, 256 },
      { a.lh, 256, a.h_whh, 256, 0, 256 } };
    gemm_block(p, 2, 0, 64, 1024, a.hlb, nullptr, 0, nullptr, 0, 0, a.lg, 1024);
    __syncthreads();
    for (int pcs = tid; pcs < 64 * 256; pcs += 256) {
      int b = pcs >> 8, j = pcs & 255;
      const float* g = a.lg + (long)b * 1024;
      float c = sigm(g[256 + j]) * a.lc[pcs] + sigm(g[j]) * tanhf(g[512 + j]);
      float h = sigm(g[768 + j]) * tanhf(c);
      a.lc[pcs] = c; a.lh[pcs] = h;
      a.hstate[((long)b * 50 + t) * 256 + j] = h;
    }
    __syncthreads();
  }
}

__global__ void k_post(const float* hiddens, const float* w1_w, const float* w1_b, float* e1) {
  int pcs = blockIdx.x * 256 + threadIdx.x;
  if (pcs < 64 * 64) {
    int b = pcs >> 6, j = pcs & 63;
    const float* ek = hiddens + (long)b * 50 * 256;
    const float* w = w1_w + (long)j * 512;
    float s = w1_b[j];
    for (int d = 0; d < 256; ++d) s += ek[d] * w[d];
    e1[pcs] = s;
  }
}

struct AlignArgs { const float* hiddens; const float* whp; const float* e1;
  const float* w1_w; const float* w2_w; const float* w2_b; float* aligned; };
__global__ void k_align(AlignArgs a) {
  int r = blockIdx.x * 256 + threadIdx.x;
  if (r < 64 * 49) {
    int b = r / 49, t = r % 49;
    const float* wr = a.whp + ((long)b * 50 + t) * 256;
    float t64[64];
    for (int j = 0; j < 64; ++j) {
      float s = a.e1[b * 64 + j];
      const float* w = a.w1_w + (long)j * 512 + 256;
      for (int d = 0; d < 256; ++d) s += wr[d] * w[d];
      t64[j] = tanhf(s);
    }
    float a0 = a.w2_b[0], a1 = a.w2_b[1];
    for (int j = 0; j < 64; ++j) { a0 += t64[j] * a.w2_w[j]; a1 += t64[j] * a.w2_w[64 + j]; }
    float m = fmaxf(a0, a1);
    float e0 = expf(a0 - m), e1v = expf(a1 - m);
    float inv = 1.f / (e0 + e1v);
    a0 = e0 * inv; a1 = e1v * inv;
    const float* ek = a.hiddens + (long)b * 50 * 256;
    float* out = a.aligned + ((long)b * 50 + t + 1) * 256;
    for (int d = 0; d < 256; ++d) out[d] = ek[d] * a0 + wr[d] * a1;
  } else if (r < 64 * 49 + 64) {
    int b = r - 64 * 49;
    const float* ek = a.hiddens + (long)b * 50 * 256;
    float* out = a.aligned + (long)b * 50 * 256;
    for (int d = 0; d < 256; ++d) out[d] = ek[d];
  }
}

__global__ void k_dconst(const float* cp, const int* dt, const float* dft_w,
                         const float* dft_b, const float* df1_b, float* cvec, float* alpha) {
  __shared__ float temb[128];
  int b = blockIdx.x, tid = threadIdx.x;
  float t = (float)dt[b];
  if (tid < 64) {
    float fr = expf(-logf(10000.f) * (float)tid / 64.f);
    float ar = t * fr;
    temb[tid] = cosf(ar); temb[64 + tid] = sinf(ar);
  }
  if (tid == 0) alpha[b] = cp[dt[b]];
  __syncthreads();
  float s = dft_b[tid] + df1_b[tid];
  const float* w = dft_w + (long)tid * 128;
  for (int k = 0; k < 128; ++k) s += temb[k] * w[k];
  cvec[b * 256 + tid] = s;
}

__global__ void k_noisy(const float* aligned, const float* noise, const float* alpha, float* noisy) {
  long i = (long)blockIdx.x * 256 + threadIdx.x;
  if (i < 819200L) {
    int b = (int)(i / (50 * 256));
    float al = alpha[b];
    noisy[i] = aligned[i] * sqrtf(al) + noise[i] * sqrtf(fmaxf(1.f - al, 0.f));
  }
}

__global__ void k_gen(const float* aligned, const float* noise, const float* pred, float* gen) {
  long i = (long)blockIdx.x * 256 + threadIdx.x;
  if (i < 819200L) gen[i] = aligned[i] + noise[i] - pred[i];
}

__global__ void k_denom(const float* cm, float* accum) {
  __shared__ float red[256];
  long i = (long)blockIdx.x * 256 + threadIdx.x;
  float v = 0.f;
  if (i < 128000L) v = 1.f - cm[i];
  red[threadIdx.x] = v;
  __syncthreads();
  for (int s = 128; s > 0; s >>= 1) {
    if (threadIdx.x < s) red[threadIdx.x] += red[threadIdx.x + s];
    __syncthreads();
  }
  if (threadIdx.x == 0) atomicAdd(&accum[1], red[0]);
}

struct FinalArgs { const float* gah; const float* fused; const float* ga2_w; const float* ga2_b;
  const float* out_w; const float* out_b; const int* lengths; const float* accum;
  float* og; float* dout; };
__global__ void __launch_bounds__(256) k_final(FinalArgs a) {
  __shared__ float ge[64 * 50];
  int tid = threadIdx.x;
  for (int pcs = tid; pcs < 3200; pcs += 256) {
    int b = pcs / 50, s = pcs % 50;
    float e = -1e30f;
    if (s < a.lengths[b]) {
      const float* g = a.gah + (long)pcs * 256;
      e = a.ga2_b[0];
      for (int n = 0; n < 256; ++n) e += g[n] * a.ga2_w[n];
    }
    ge[pcs] = e;
  }
  __syncthreads();
  if (tid < 64) {
    float* r = ge + tid * 50;
    float m = -1e30f; for (int s = 0; s < 50; ++s) m = fmaxf(m, r[s]);
    float ss = 0.f; for (int s = 0; s < 50; ++s) { float t = expf(r[s] - m); r[s] = t; ss += t; }
    float inv = 1.f / ss; for (int s = 0; s < 50; ++s) r[s] *= inv;
  }
  __syncthreads();
  for (int pcs = tid; pcs < 64 * 256; pcs += 256) {
    int b = pcs >> 8, d = pcs & 255;
    float o = 0.f;
    const float* f = a.fused + (long)b * 50 * 256 + d;
    const float* w = ge + b * 50;
    for (int s = 0; s < 50; ++s) o += w[s] * f[(long)s * 256];
    a.og[pcs] = o;
  }
  __syncthreads();
  if (tid < 128) {
    int b = tid >> 1, k = tid & 1;
    float r = a.out_b[k];
    const float* o = a.og + (long)b * 256;
    const float* w = a.out_w + k * 256;
    for (int d = 0; d < 256; ++d) r += o[d] * w[d];
    a.dout[b * 2 + k] = r;
  }
  if (tid == 0) a.dout[128] = 1.f - a.accum[0] / a.accum[1];
}

// ---------------- host launch ----------------
extern "C" void kernel_launch(void* const* d_in, const int* in_sizes, int n_in,
                              void* d_out, int out_size, void* d_ws, size_t ws_size,
                              hipStream_t stream) {
  (void)in_sizes; (void)n_in; (void)out_size; (void)ws_size;
  const int*   input_seqs = (const int*)d_in[0];
  const int*   lengths    = (const int*)d_in[1];
  const float* stime      = (const float*)d_in[2];
  const float* code_mask  = (const float*)d_in[3];
  const float* gumbel_p   = (const float*)d_in[4];
  const float* gumbel_v   = (const float*)d_in[5];
  const int*   diff_t     = (const int*)d_in[6];
  const float* noise      = (const float*)d_in[7];
  const float* emb   = (const float*)d_in[8];
  const float* tgt   = (const float*)d_in[9];
  const float* tl_w  = (const float*)d_in[10]; const float* tl_b  = (const float*)d_in[11];
  const float* tl2_w = (const float*)d_in[12]; const float* tl2_b = (const float*)d_in[13];
  const float* cs1_w = (const float*)d_in[14]; const float* cs1_b = (const float*)d_in[15];
  const float* cs2_w = (const float*)d_in[16]; const float* cs2_b = (const float*)d_in[17];
  const float* rr_wih= (const float*)d_in[18]; const float* rr_whh= (const float*)d_in[19];
  const float* vs1_w = (const float*)d_in[20]; const float* vs1_b = (const float*)d_in[21];
  const float* vs2_w = (const float*)d_in[22]; const float* vs2_b = (const float*)d_in[23];
  const float* cq_w  = (const float*)d_in[24]; const float* cq_b  = (const float*)d_in[25];
  const float* fc_w  = (const float*)d_in[26]; const float* fc_b  = (const float*)d_in[27];
  const float* gru_wih = (const float*)d_in[28]; const float* gru_whh = (const float*)d_in[29];
  const float* t1_w  = (const float*)d_in[30]; const float* t1_b  = (const float*)d_in[31];
  const float* t2_w  = (const float*)d_in[32]; const float* t2_b  = (const float*)d_in[33];
  const float* tu_w  = (const float*)d_in[34]; const float* tu_b  = (const float*)d_in[35];
  const float* mem_w = (const float*)d_in[36]; const float* mem_b = (const float*)d_in[37];
  const float* mem_g = (const float*)d_in[38]; const float* mem_be= (const float*)d_in[39];
  const float* st_w  = (const float*)d_in[40]; const float* st_b  = (const float*)d_in[41];
  const float* st_g  = (const float*)d_in[42]; const float* st_be = (const float*)d_in[43];
  const float* ff1_w = (const float*)d_in[44]; const float* ff1_b = (const float*)d_in[45];
  const float* ff2_w = (const float*)d_in[46]; const float* ff2_b = (const float*)d_in[47];
  const float* ln_g  = (const float*)d_in[48]; const float* ln_b  = (const float*)d_in[49];
  const float* ln2_g = (const float*)d_in[50]; const float* ln2_b = (const float*)d_in[51];
  const float* ga1_w = (const float*)d_in[52]; const float* ga1_b = (const float*)d_in[53];
  const float* ga2_w = (const float*)d_in[54]; const float* ga2_b = (const float*)d_in[55];
  const float* out_w = (const float*)d_in[56]; const float* out_b = (const float*)d_in[57];
  const float* fuse_w= (const float*)d_in[58]; const float* fuse_b= (const float*)d_in[59];
  const float* hl_wih= (const float*)d_in[60]; const float* hl_whh= (const float*)d_in[61];
  const float* hl_bih= (const float*)d_in[62]; const float* hl_bhh= (const float*)d_in[63];
  const float* whk_w = (const float*)d_in[64]; const float* whk_b = (const float*)d_in[65];
  const float* w1_w  = (const float*)d_in[66]; const float* w1_b  = (const float*)d_in[67];
  const float* w2_w  = (const float*)d_in[68]; const float* w2_b  = (const float*)d_in[69];
  const float* df1_w = (const float*)d_in[70]; const float* df1_b = (const float*)d_in[71];
  const float* dft_w = (const float*)d_in[72]; const float* dft_b = (const float*)d_in[73];
  const float* df2_w = (const float*)d_in[74]; const float* df2_b = (const float*)d_in[75];
  const float* betas = (const float*)d_in[76];

  size_t off = 0;
  auto A16 = [&](long n) { off = (off + 255) & ~(size_t)255; _Float16* p = (_Float16*)((char*)d_ws + off); off += (size_t)n * 2; return p; };
  auto A32 = [&](long n) { off = (off + 255) & ~(size_t)255; float* p = (float*)((char*)d_ws + off); off += (size_t)n * 4; return p; };

  // f16 weights
  _Float16* h_cs1  = A16(256L*512);  _Float16* h_rrwih = A16(128L*256);
  _Float16* h_rrwhh= A16(128L*32);   _Float16* h_vs1   = A16(256L*800);
  _Float16* h_cq   = A16(256L*544);  _Float16* h_fc    = A16(256L*256);
  _Float16* h_gwih = A16(768L*256);  _Float16* h_gwhh  = A16(768L*256);
  _Float16* h_mem  = A16(256L*256);  _Float16* h_st    = A16(256L*256);
  _Float16* h_ff1  = A16(1024L*256); _Float16* h_ff2   = A16(256L*1024);
  _Float16* h_hwih = A16(1024L*256); _Float16* h_hwhh  = A16(1024L*256);
  _Float16* h_whk  = A16(256L*256);  _Float16* h_df1   = A16(256L*256);
  _Float16* h_df2  = A16(256L*256);  _Float16* h_fuse  = A16(256L*512);
  _Float16* h_ga1  = A16(256L*576);  _Float16* h_tu    = A16(256L*64);
  // f32 scratch
  float* cp      = A32(1000);
  float* tlin    = A32(256);   float* tlin2   = A32(256);
  float* csconst = A32(256);   float* vconst  = A32(256);
  float* cqconst = A32(256);   float* gaconst = A32(256);
  float* hlb     = A32(1024);  float* accum   = A32(8);
  float* alpha   = A32(64);    float* cvec    = A32(64L*256);
  float* e1      = A32(64L*64);float* pm      = A32(20001);
  float* tf      = A32(3200L*64);  float* tf2   = A32(3200L*64);
  float* time_enc= A32(819200);    float* selected = A32(819200);
  float* fol     = A32(102400);    float* hiddens  = A32(819200);
  float* memrows = A32(819200);    float* hstate   = A32(819200);
  float* whp     = A32(819200);    float* aligned  = A32(819200);
  float* noisy   = A32(819200);    float* dfh      = A32(819200);
  float* pred    = A32(819200);    float* gen      = A32(819200);
  float* fused   = A32(819200);    float* gah      = A32(819200);
  float* og      = A32(64L*256);
  float* hx  = A32(64L*256); float* h1 = A32(64L*256); float* cqb = A32(64L*256);
  float* zb  = A32(64L*256); float* zf = A32(64L*256);
  float* gi  = A32(64L*768); float* gh = A32(64L*768);
  float* shb = A32(64L*256); float* shl= A32(64L*256);
  float* hxn = A32(64L*256); float* hxn2=A32(64L*256); float* f1 = A32(64L*1024);
  float* fh  = A32(64L*32);  float* fcs= A32(64L*32);  float* fg = A32(64L*128);
  float* lh  = A32(64L*256); float* lc = A32(64L*256); float* lg = A32(64L*1024);

  auto H = [&](const float* s, _Float16* d, long n) {
    k_h<<<dim3((unsigned)((n + 255) / 256)), dim3(256), 0, stream>>>(s, d, n);
  };
  H(cs1_w, h_cs1, 256L*512); H(rr_wih, h_rrwih, 128L*256); H(rr_whh, h_rrwhh, 128L*32);
  H(vs1_w, h_vs1, 256L*800); H(cq_w, h_cq, 256L*544);      H(fc_w, h_fc, 256L*256);
  H(gru_wih, h_gwih, 768L*256); H(gru_whh, h_gwhh, 768L*256);
  H(mem_w, h_mem, 256L*256); H(st_w, h_st, 256L*256);
  H(ff1_w, h_ff1, 1024L*256); H(ff2_w, h_ff2, 256L*1024);
  H(hl_wih, h_hwih, 1024L*256); H(hl_whh, h_hwhh, 1024L*256);
  H(whk_w, h_whk, 256L*256); H(df1_w, h_df1, 256L*256); H(df2_w, h_df2, 256L*256);
  H(fuse_w, h_fuse, 256L*512); H(ga1_w, h_ga1, 256L*576); H(tu_w, h_tu, 256L*64);

  { PrepArgs a = { betas, tgt, tl_w, tl_b, tl2_w, tl2_b, cs1_w, cs1_b, vs1_w, vs1_b,
                   cq_w, cq_b, ga1_w, ga1_b, hl_bih, hl_bhh,
                   cp, tlin, tlin2, csconst, vconst, cqconst, gaconst, hlb, accum };
    k_prep<<<1, 256, 0, stream>>>(a); }

  k_time<<<800, 256, 0, stream>>>(stime, t1_w, t1_b, t2_w, t2_b, tf, tf2);

  auto G = [&](int np, Part p0, Part p1, Part p2, long rows, int N, const float* bias,
               const float* addv, long ldadd, const float* rowvec, int rowdiv, int act,
               float* C, long ldc) {
    GemmArgs g; g.p[0]=p0; g.p[1]=p1; g.p[2]=p2; g.np=np; g.rows=rows; g.N=N; g.bias=bias;
    g.addv=addv; g.ldadd=ldadd; g.rowvec=rowvec; g.rowdiv=rowdiv; g.act=act; g.C=C; g.ldc=ldc;
    k_gemm<<<dim3((unsigned)((rows + 63) / 64)), dim3(256), 0, stream>>>(g);
  };
  Part PZ = { nullptr, 0, nullptr, 0, 0, 0 };

  // time_enc = tf @ tu.T + tu_b
  G(1, Part{tf, 64, h_tu, 64, 0, 64}, PZ, PZ, 3200, 256, tu_b,
    nullptr, 0, nullptr, 0, 0, time_enc, 256);

  { PmArgs a = { emb, h_cs1, csconst, cs2_w, cs2_b, gumbel_p, pm };
    k_pm<<<626, 256, 0, stream>>>(a); }

  k_selected<<<3200, 256, 0, stream>>>(input_seqs, emb, pm, time_enc, selected);

  { FolArgs a = { selected, h_rrwih, h_rrwhh, fh, fcs, fg, fol };
    k_fol<<<1, 256, 0, stream>>>(a); }

  { ScanArgs a;
    a.emb = emb; a.idx = input_seqs; a.gum_v = gumbel_v; a.pm = pm;
    a.selected = selected; a.fol = fol; a.time_enc = time_enc;
    a.h_vs1 = h_vs1; a.h_cq = h_cq; a.h_fc = h_fc;
    a.h_gru_wih = h_gwih; a.h_gru_whh = h_gwhh;
    a.h_mem = h_mem; a.h_st = h_st; a.h_ff1 = h_ff1; a.h_ff2 = h_ff2;
    a.vconst = vconst; a.cqconst = cqconst;
    a.vs2_w = vs2_w; a.vs2_b = vs2_b;
    a.fc_b = fc_b; a.mem_b = mem_b; a.mem_g = mem_g; a.mem_be = mem_be;
    a.st_b = st_b; a.st_g = st_g; a.st_be = st_be;
    a.ff1_b = ff1_b; a.ff2_b = ff2_b;
    a.ln_g = ln_g; a.ln_b = ln_b; a.ln2_g = ln2_g; a.ln2_b = ln2_b;
    a.hx = hx; a.h1 = h1; a.cq = cqb; a.z = zb; a.zf = zf; a.gi = gi; a.gh = gh;
    a.sh = shb; a.shl = shl; a.hxn = hxn; a.hxn2 = hxn2; a.f1 = f1;
    a.hiddens = hiddens; a.memrows = memrows; a.accum = accum;
    k_scan<<<1, 256, 0, stream>>>(a); }

  k_post<<<16, 256, 0, stream>>>(hiddens, w1_w, w1_b, e1);

  { HlArgs a = { hiddens, h_hwih, h_hwhh, hlb, lh, lc, lg, hstate };
    k_hlstm<<<1, 256, 0, stream>>>(a); }

  // whp = hiddenstate @ whk.T + whk_b
  G(1, Part{hstate, 256, h_whk, 256, 0, 256}, PZ, PZ, 3200, 256, whk_b,
    nullptr, 0, nullptr, 0, 0, whp, 256);

  { AlignArgs a = { hiddens, whp, e1, w1_w, w2_w, w2_b, aligned };
    k_align<<<13, 256, 0, stream>>>(a); }

  k_dconst<<<64, 256, 0, stream>>>(cp, diff_t, dft_w, dft_b, df1_b, cvec, alpha);
  k_noisy<<<3200, 256, 0, stream>>>(aligned, noise, alpha, noisy);

  // h = relu(noisy @ df1.T + cvec[b])
  G(1, Part{noisy, 256, h_df1, 256, 0, 256}, PZ, PZ, 3200, 256, nullptr,
    nullptr, 0, cvec, 50, 1, dfh, 256);
  // pred = h @ df2.T + df2_b
  G(1, Part{dfh, 256, h_df2, 256, 0, 256}, PZ, PZ, 3200, 256, df2_b,
    nullptr, 0, nullptr, 0, 0, pred, 256);
  k_gen<<<3200, 256, 0, stream>>>(aligned, noise, pred, gen);

  // fused = [hiddens, GEN] @ fuse.T + fuse_b
  G(2, Part{hiddens, 256, h_fuse, 512, 0, 256}, Part{gen, 256, h_fuse, 512, 256, 256}, PZ,
    3200, 256, fuse_b, nullptr, 0, nullptr, 0, 0, fused, 256);

  // gah = tanh([fused, tf2, tgt] @ ga1.T + ga1_b)
  G(2, Part{fused, 256, h_ga1, 576, 0, 256}, Part{tf2, 64, h_ga1, 576, 256, 64}, PZ,
    3200, 256, gaconst, nullptr, 0, nullptr, 0, 2, gah, 256);

  k_denom<<<500, 256, 0, stream>>>(code_mask, accum);

  { FinalArgs a = { gah, fused, ga2_w, ga2_b, out_w, out_b, lengths, accum,
                    og, (float*)d_out };
    k_final<<<1, 256, 0, stream>>>(a); }
}
